// SurealCaptionModel_75316546503230
// MI455X (gfx1250) — matrix operand order repaired and checked by reading, not compiled, not run on hardware
//
#include <hip/hip_runtime.h>
#include <hip/hip_bf16.h>
#include <math.h>

// ---------------------------------------------------------------------------
// LSTM captioner for MI455X (gfx1250, wave32, WMMA).
// All GEMMs: bf16 WMMA (v_wmma_f32_16x16x32_bf16, f32 accum).
// Weights are pre-swizzled into WMMA fragment-packed tiles so each B fragment
// is one contiguous 32B load per lane (no scalar u16 gathers).
// Sequential recurrence: persistent kernel, one workgroup per 16-row batch
// tile, h/c state + gates in LDS, packed W_hh (2MB bf16) L2-resident.
// The W_hh pointer is laundered through asm each time-step (blocks LICM
// hoist/spill) but reloaded via an address_space(1) pointer so the loads
// stay global_load_b128 (not flat), keeping LOADcnt/DScnt decoupled.
// ---------------------------------------------------------------------------

typedef __attribute__((ext_vector_type(16))) __bf16   v16bf;
typedef __attribute__((ext_vector_type(8)))  float    v8f;
typedef __attribute__((ext_vector_type(8)))  unsigned v8u;
typedef __attribute__((ext_vector_type(4)))  unsigned v4u;

typedef const __attribute__((address_space(1))) v8u* gv8u_ptr;

static __device__ __forceinline__ __bf16 f2bf(float f) {
    unsigned u = __builtin_bit_cast(unsigned, f);
    unsigned r = u + 0x7FFFu + ((u >> 16) & 1u);   // round-to-nearest-even
    unsigned short h = (unsigned short)(r >> 16);
    return __builtin_bit_cast(__bf16, h);
}

// A fragment: 16x32 (MxK) bf16 from row-major A (ld = leading dim, mult of 8).
// ISA layout: lanes 0-15 hold M=lane, K={0..7,16..23}; lanes 16-31 hold
// M=lane-16, K={8..15,24..31}.  Two contiguous 16B chunks per lane.
static __device__ __forceinline__ v16bf load_a_frag(const __bf16* A, int ld,
                                                    int m0, int k0, int lane) {
    int row = m0 + (lane & 15);
    int kh  = ((lane >> 4) << 3);          // 0 or 8
    const __bf16* p = A + (size_t)row * ld + k0 + kh;
    v4u lo = *reinterpret_cast<const v4u*>(p);        // K = kh+0..7
    v4u hi = *reinterpret_cast<const v4u*>(p + 16);   // K = kh+16..23
    v8u c  = __builtin_shufflevector(lo, hi, 0, 1, 2, 3, 4, 5, 6, 7);
    return __builtin_bit_cast(v16bf, c);
}

// B fragment from fragment-packed weights: tile (kt,nt) stored lane-major,
// 16 contiguous bf16 (32B) per lane.  One vector load, fully coalesced.
static __device__ __forceinline__ v16bf load_b_frag_packed(const __bf16* P,
                                                           int ktiles, int kt,
                                                           int nt, int lane) {
    const __bf16* p = P + ((((size_t)nt * ktiles + kt) << 5) + lane) * 16;
    v8u raw = *reinterpret_cast<const v8u*>(p);
    return __builtin_bit_cast(v16bf, raw);
}

// Same, but from a laundered integer base cast to global address space so the
// compiler emits global_load_b128 (not flat) yet cannot hoist across the
// laundering asm.
static __device__ __forceinline__ v16bf load_b_frag_packed_g(
        unsigned long long base, int ktiles, int kt, int nt, int lane) {
    unsigned long long off =
        ((((unsigned long long)(nt * ktiles + kt)) << 5) + (unsigned)lane) * 32ull;
    gv8u_ptr p = (gv8u_ptr)(base + off);
    v8u raw = *p;
    return __builtin_bit_cast(v16bf, raw);
}

static __device__ __forceinline__ v8f wmma_bf16(v16bf a, v16bf b, v8f c) {
    return __builtin_amdgcn_wmma_f32_16x16x32_bf16(
        false, a, false, b, (short)0, c, false, false);
}

// ---------------------------------------------------------------------------
// Conversion / prep kernels
// ---------------------------------------------------------------------------
__global__ void cast_bf16_kernel(const float* __restrict__ src,
                                 __bf16* __restrict__ dst, int n) {
    for (int i = blockIdx.x * blockDim.x + threadIdx.x; i < n;
         i += gridDim.x * blockDim.x)
        dst[i] = f2bf(src[i]);
}

// Pack f32 weight W[N][ld] (cols col0..col0+K-1) into WMMA-fragment tiles:
// dst[((nt*ktiles + kt)*32 + lane)*16 + e] = bf16(W[nt*16 + (lane&15)]
//                                                 [col0 + kt*32 + (lane>>4)*8 + (e&7) + (e>>3)*16])
__global__ void pack_weight_kernel(const float* __restrict__ src,
                                   __bf16* __restrict__ dst,
                                   int N, int K, int ld, int col0) {
    int total  = N * K;
    int ktiles = K >> 5;
    for (int idx = blockIdx.x * blockDim.x + threadIdx.x; idx < total;
         idx += gridDim.x * blockDim.x) {
        int e    = idx & 15;
        int lane = (idx >> 4) & 31;
        int tile = idx >> 9;
        int kt   = tile % ktiles;
        int nt   = tile / ktiles;
        int n = (nt << 4) + (lane & 15);
        int k = (kt << 5) + ((lane >> 4) << 3) + (e & 7) + ((e >> 3) << 4);
        dst[idx] = f2bf(src[(size_t)n * ld + col0 + k]);
    }
}

__global__ void bias_sum_kernel(const float* __restrict__ a,
                                const float* __restrict__ b,
                                float* __restrict__ out, int n) {
    int i = blockIdx.x * blockDim.x + threadIdx.x;
    if (i < n) out[i] = a[i] + b[i];
}

// Gather caption embeddings into time-major bf16: out[(t*128+b)*512 + e]
__global__ void gather_emb_kernel(const int* __restrict__ captions,
                                  const float* __restrict__ table,
                                  __bf16* __restrict__ out) {
    int idx = blockIdx.x * blockDim.x + threadIdx.x;  // 3968*512 total
    if (idx >= 3968 * 512) return;
    int e = idx & 511;
    int m = idx >> 9;
    int b = m & 127;
    int t = m >> 7;                                    // 0..30 (drops last col)
    int tok = captions[b * 32 + t];
    out[idx] = f2bf(table[(size_t)tok * 512 + e]);
}

// ---------------------------------------------------------------------------
// Generic bf16 WMMA GEMM: C[M,N] = A[M,K](bf16) x Bp(packed W^T) (+bias[N])
// (+addend[(row&127)][N]).  Each wave: one 16-row M tile x FOUR 16-col N
// tiles.  Per k-step: issue all 4 B fragments + the A fragment first, then
// run the 4 WMMAs (loads clause together, waits stagger).  8 waves / block.
// OUT_MODE: 0 = f32 store, 1 = bf16 store.
// OUT_PERM: store row (t*128+b) at output row (b*31+t)  [final logits].
// ---------------------------------------------------------------------------
template<int OUT_MODE, bool HAS_BIAS, bool HAS_ADDEND, bool OUT_PERM>
__global__ __launch_bounds__(256)
void gemm_bf16_wmma(const __bf16* __restrict__ A, const __bf16* __restrict__ Bp,
                    const float* __restrict__ bias,
                    const float* __restrict__ addend,
                    float* __restrict__ Cf, __bf16* __restrict__ Cb,
                    int M, int N, int K) {
    const int lane    = threadIdx.x & 31;
    const int wave    = threadIdx.x >> 5;
    const int tiles_n = N >> 4;
    const int ktiles  = K >> 5;
    const int tn0     = (blockIdx.y * 8 + wave) * 4;
    if (tn0 >= tiles_n) return;
    const int nvalid = (tiles_n - tn0 < 4) ? (tiles_n - tn0) : 4;
    const int m0 = blockIdx.x << 4;

    v8f acc[4] = {};
    for (int kt = 0; kt < ktiles; ++kt) {
        v16bf bfr[4];
#pragma unroll
        for (int q = 0; q < 4; ++q)
            if (q < nvalid)       // uniform per wave -> EXEC stays all-ones
                bfr[q] = load_b_frag_packed(Bp, ktiles, kt, tn0 + q, lane);
        v16bf a = load_a_frag(A, K, m0, kt << 5, lane);
#pragma unroll
        for (int q = 0; q < 4; ++q)
            if (q < nvalid)
                acc[q] = wmma_bf16(a, bfr[q], acc[q]);
    }

#pragma unroll
    for (int q = 0; q < 4; ++q) {
        if (q >= nvalid) break;
        const int col = ((tn0 + q) << 4) + (lane & 15);
        const float bc = HAS_BIAS ? bias[col] : 0.0f;
#pragma unroll
        for (int r = 0; r < 8; ++r) {
            int row = m0 + r + ((lane >> 4) << 3);
            float v = acc[q][r] + bc;
            if (HAS_ADDEND) v += addend[(size_t)(row & 127) * N + col];
            if (OUT_MODE == 0) {
                size_t orow = OUT_PERM ? (size_t)((row & 127) * 31 + (row >> 7))
                                       : (size_t)row;
                Cf[orow * (size_t)N + col] = v;
            } else {
                Cb[(size_t)row * N + col] = f2bf(v);
            }
        }
    }
}

// ---------------------------------------------------------------------------
// Persistent LSTM scan. 8 blocks x 1024 threads (32 waves), block owns 16
// batch rows for all 31 steps. LDS: h(bf16) 16K + c(f32) 32K + gates 128K.
// Per step: gates[16,2048] = xg[t] + h @ Whh^T  (WMMA, A fragment from LDS
// reused across the wave's 4 gate N-tiles), then LSTM cell elementwise.
// ---------------------------------------------------------------------------
__global__ __launch_bounds__(1024)
void lstm_scan_kernel(const float* __restrict__ xg,    // [31][128][2048]
                      const __bf16* __restrict__ h0,   // [128][512]
                      const float* __restrict__ c0,    // [128][512]
                      const __bf16* __restrict__ Whhp, // packed, ktiles=16
                      __bf16* __restrict__ hs) {       // [31][128][512]
    extern __shared__ char smem_raw[];
    __bf16* h_sm = (__bf16*)smem_raw;                       // 16*512*2
    float*  c_sm = (float*)(smem_raw + 16384);              // 16*512*4
    float*  g_sm = (float*)(smem_raw + 16384 + 32768);      // 16*2048*4

    const int tid  = threadIdx.x;
    const int lane = tid & 31;
    const int wave = tid >> 5;            // 0..31
    const int m0   = blockIdx.x << 4;     // global batch row base

#pragma unroll
    for (int it = 0; it < 8; ++it) {
        int idx = tid + it * 1024;        // 0..8191
        int lr = idx >> 9, j = idx & 511;
        h_sm[idx] = h0[(size_t)(m0 + lr) * 512 + j];
        c_sm[idx] = c0[(size_t)(m0 + lr) * 512 + j];
    }
    __syncthreads();

    unsigned long long wp = (unsigned long long)Whhp;

#pragma unroll 1
    for (int t = 0; t < 31; ++t) {
        // Optimization barrier: redefine the weight base every step so the 64
        // invariant B-fragment loads stay inside the loop (no hoist/spill).
        asm volatile("" : "+s"(wp));

        const float* xgt = xg + (size_t)t * 128 * 2048;

        // Wave handles gate N-tiles {wave, wave+32, wave+64, wave+96}.
        v8f acc[4];
#pragma unroll
        for (int q = 0; q < 4; ++q) {
            int coln = ((wave + q * 32) << 4) + (lane & 15);
#pragma unroll
            for (int r = 0; r < 8; ++r) {
                int lr = r + ((lane >> 4) << 3);
                acc[q][r] = xgt[(size_t)(m0 + lr) * 2048 + coln];
            }
        }
#pragma unroll 4
        for (int kt = 0; kt < 16; ++kt) {
            v16bf bfr[4];
#pragma unroll
            for (int q = 0; q < 4; ++q)
                bfr[q] = load_b_frag_packed_g(wp, 16, kt, wave + q * 32, lane);
            v16bf a = load_a_frag(h_sm, 512, 0, kt << 5, lane);  // from LDS
#pragma unroll
            for (int q = 0; q < 4; ++q)
                acc[q] = wmma_bf16(a, bfr[q], acc[q]);
        }
#pragma unroll
        for (int q = 0; q < 4; ++q) {
            int coln = ((wave + q * 32) << 4) + (lane & 15);
#pragma unroll
            for (int r = 0; r < 8; ++r) {
                int lr = r + ((lane >> 4) << 3);
                g_sm[lr * 2048 + coln] = acc[q][r];
            }
        }
        __syncthreads();

        // LSTM cell elementwise: 16 rows x 512 hidden, 8 per thread
#pragma unroll
        for (int it = 0; it < 8; ++it) {
            int idx = tid + it * 1024;
            int lr = idx >> 9, j = idx & 511;
            float gi = g_sm[lr * 2048 + j];
            float gf = g_sm[lr * 2048 + 512 + j];
            float gg = g_sm[lr * 2048 + 1024 + j];
            float go = g_sm[lr * 2048 + 1536 + j];
            float i = 1.0f / (1.0f + __expf(-gi));
            float f = 1.0f / (1.0f + __expf(-gf));
            float g = tanhf(gg);
            float o = 1.0f / (1.0f + __expf(-go));
            float c = f * c_sm[idx] + i * g;
            float h = o * tanhf(c);
            c_sm[idx] = c;
            __bf16 hb = f2bf(h);
            h_sm[idx] = hb;
            hs[((size_t)t * 128 + (m0 + lr)) * 512 + j] = hb;
        }
        __syncthreads();
    }
}

// ---------------------------------------------------------------------------
extern "C" void kernel_launch(void* const* d_in, const int* in_sizes, int n_in,
                              void* d_out, int out_size, void* d_ws, size_t ws_size,
                              hipStream_t stream) {
    (void)in_sizes; (void)n_in; (void)out_size; (void)ws_size;

    const float* feat      = (const float*)d_in[0];   // [128,2048]
    const int*   captions  = (const int*)  d_in[1];   // [128,32]
    const float* emb_table = (const float*)d_in[2];   // [10000,512]
    const float* W_init_h  = (const float*)d_in[3];   // [512,2048]
    const float* b_init_h  = (const float*)d_in[4];
    const float* W_init_c  = (const float*)d_in[5];
    const float* b_init_c  = (const float*)d_in[6];
    const float* W_proj    = (const float*)d_in[7];   // [512,2560]
    const float* b_proj    = (const float*)d_in[8];
    const float* W_ih      = (const float*)d_in[9];   // [2048,512]
    const float* b_ih      = (const float*)d_in[10];
    const float* W_hh      = (const float*)d_in[11];  // [2048,512]
    const float* b_hh      = (const float*)d_in[12];
    const float* W_out     = (const float*)d_in[13];  // [10000,512]
    const float* b_out     = (const float*)d_in[14];
    float* out = (float*)d_out;                        // [128,31,10000]

    char* ws = (char*)d_ws;
    size_t off = 0;
    auto take = [&](size_t bytes) -> void* {
        void* p = ws + off;
        off += (bytes + 255) & ~(size_t)255;
        return p;
    };

    __bf16* feat_bf   = (__bf16*)take(128u * 2048 * 2);
    __bf16* Wp_init_h = (__bf16*)take(512u * 2048 * 2);   // packed
    __bf16* Wp_init_c = (__bf16*)take(512u * 2048 * 2);
    __bf16* Wp_proj1  = (__bf16*)take(512u * 2048 * 2);   // feat part, K=2048
    __bf16* Wp_proj2  = (__bf16*)take(512u * 512 * 2);    // emb part,  K=512
    __bf16* Wp_ih     = (__bf16*)take(2048u * 512 * 2);
    __bf16* Wp_hh     = (__bf16*)take(2048u * 512 * 2);
    __bf16* Wp_out    = (__bf16*)take(10000u * 512 * 2);
    float*  bias_sum  = (float*) take(2048u * 4);
    __bf16* emb_bf    = (__bf16*)take(3968u * 512 * 2);   // [t*128+b][512]
    __bf16* h0_bf     = (__bf16*)take(128u * 512 * 2);
    float*  c0_f      = (float*) take(128u * 512 * 4);
    float*  projfeat  = (float*) take(128u * 512 * 4);
    __bf16* x_all_bf  = (__bf16*)take(3968u * 512 * 2);
    float*  xg        = (float*) take((size_t)3968 * 2048 * 4); // [31][128][2048]
    __bf16* hs_bf     = (__bf16*)take(3968u * 512 * 2);   // [31][128][512]

    // --- precision conversion + fragment packing (cheap, one pass each) ---
    cast_bf16_kernel<<<1024, 256, 0, stream>>>(feat, feat_bf, 128 * 2048);
    pack_weight_kernel<<<4096, 256, 0, stream>>>(W_init_h, Wp_init_h, 512, 2048, 2048, 0);
    pack_weight_kernel<<<4096, 256, 0, stream>>>(W_init_c, Wp_init_c, 512, 2048, 2048, 0);
    pack_weight_kernel<<<4096, 256, 0, stream>>>(W_proj,   Wp_proj1,  512, 2048, 2560, 0);
    pack_weight_kernel<<<1024, 256, 0, stream>>>(W_proj,   Wp_proj2,  512,  512, 2560, 2048);
    pack_weight_kernel<<<4096, 256, 0, stream>>>(W_ih,     Wp_ih,    2048,  512,  512, 0);
    pack_weight_kernel<<<4096, 256, 0, stream>>>(W_hh,     Wp_hh,    2048,  512,  512, 0);
    pack_weight_kernel<<<8192, 256, 0, stream>>>(W_out,    Wp_out,  10000,  512,  512, 0);
    bias_sum_kernel<<<8, 256, 0, stream>>>(b_ih, b_hh, bias_sum, 2048);
    gather_emb_kernel<<<(3968 * 512 + 255) / 256, 256, 0, stream>>>(
        captions, emb_table, emb_bf);

    // --- h0 (bf16) / c0 (f32): [128,512] = feat @ W_init^T + b ---
    gemm_bf16_wmma<1, true, false, false><<<dim3(8, 1), 256, 0, stream>>>(
        feat_bf, Wp_init_h, b_init_h, nullptr, nullptr, h0_bf, 128, 512, 2048);
    gemm_bf16_wmma<0, true, false, false><<<dim3(8, 1), 256, 0, stream>>>(
        feat_bf, Wp_init_c, b_init_c, nullptr, c0_f, nullptr, 128, 512, 2048);

    // --- x_all = [feat|emb] @ W_proj^T + b_proj, split along K ---
    // projfeat = feat @ W_proj[:, :2048]^T + b_proj            (f32 [128,512])
    gemm_bf16_wmma<0, true, false, false><<<dim3(8, 1), 256, 0, stream>>>(
        feat_bf, Wp_proj1, b_proj, nullptr, projfeat, nullptr, 128, 512, 2048);
    // x_all[t*128+b] = emb @ W_proj[:, 2048:]^T + projfeat[b]  (bf16)
    gemm_bf16_wmma<1, false, true, false><<<dim3(248, 1), 256, 0, stream>>>(
        emb_bf, Wp_proj2, nullptr, projfeat, nullptr, x_all_bf, 3968, 512, 512);

    // --- xg_all = x_all @ W_ih^T + (b_ih + b_hh)   (f32 [31][128][2048]) ---
    gemm_bf16_wmma<0, true, false, false><<<dim3(248, 4), 256, 0, stream>>>(
        x_all_bf, Wp_ih, bias_sum, nullptr, xg, nullptr, 3968, 2048, 512);

    // --- sequential scan: persistent, 8 batch tiles of 16 rows ---
    lstm_scan_kernel<<<8, 1024, 16384 + 32768 + 131072, stream>>>(
        xg, h0_bf, c0_f, Wp_hh, hs_bf);

    // --- logits = hs @ W_out^T + b_out, stored permuted to [b][t][v] ---
    gemm_bf16_wmma<0, true, false, true><<<dim3(248, 20), 256, 0, stream>>>(
        hs_bf, Wp_out, b_out, nullptr, out, nullptr, 3968, 10000, 512);
}
